// FlashAttention2_35485019800164
// MI455X (gfx1250) — matrix operand order repaired and checked by reading, not compiled, not run on hardware
//
#include <hip/hip_runtime.h>
#include <hip/hip_bf16.h>

#define EMBED 2048
#define HEADS 16
#define HDIM  128
#define SEQ   2048
#define BATCH 2
#define MTOT  (BATCH*SEQ)   // 4096 tokens

typedef __attribute__((ext_vector_type(16))) _Float16 v16h;
typedef __attribute__((ext_vector_type(8)))  _Float16 v8h;
typedef __attribute__((ext_vector_type(4)))  _Float16 v4h;
typedef __attribute__((ext_vector_type(8)))  float    v8f;
typedef __attribute__((ext_vector_type(4)))  int      v4i;

#define AS1 __attribute__((address_space(1)))
#define AS3 __attribute__((address_space(3)))

#if defined(__has_builtin)
#if __has_builtin(__builtin_amdgcn_global_load_async_to_lds_b128) && \
    __has_builtin(__builtin_amdgcn_s_wait_asynccnt)
#define USE_ASYNC_LDS 1
#endif
#endif
#ifndef USE_ASYNC_LDS
#define USE_ASYNC_LDS 0
#endif

static __device__ __forceinline__ v16h pack16(v8h lo, v8h hi) {
  union { v16h w; v8h h[2]; } u;
  u.h[0] = lo; u.h[1] = hi;
  return u.w;
}

// A-matrix fragment (16x32 f16): lanes 0-15 = rows, K chunks {8*hi..+7} and {16+8*hi..+7}
static __device__ __forceinline__ v16h afrag(const _Float16* row0, int lrow, int hi, int stride) {
  const _Float16* p = row0 + lrow * stride;
  return pack16(*(const v8h*)(p + 8 * hi), *(const v8h*)(p + 16 + 8 * hi));
}
// B-matrix fragment (32x16 f16): lanes 0-15 = cols, K chunks {8*hi..+7} and {16+8*hi..+7}
static __device__ __forceinline__ v16h bfrag(const _Float16* row0, int lrow, int hi, int stride) {
  const _Float16* p = row0 + lrow * stride;
  return pack16(*(const v8h*)(p + 8 * hi), *(const v8h*)(p + 16 + 8 * hi));
}

// ---------------------------------------------------------------------------
// fp32 -> f16 elementwise convert (for x)
// ---------------------------------------------------------------------------
__global__ __launch_bounds__(256)
void convert_f32_f16_kernel(const float* __restrict__ src, _Float16* __restrict__ dst, int n) {
  int i = (blockIdx.x * 256 + threadIdx.x) * 4;
  if (i < n) {
    float4 v = *(const float4*)&src[i];
    v4h pk;
    pk[0] = (_Float16)v.x; pk[1] = (_Float16)v.y;
    pk[2] = (_Float16)v.z; pk[3] = (_Float16)v.w;
    *(v4h*)&dst[i] = pk;
  }
}

// ---------------------------------------------------------------------------
// W[K,N] fp32 -> Wt[N,K] f16 (LDS tiled transpose, 64x64 tiles)
// ---------------------------------------------------------------------------
__global__ __launch_bounds__(256)
void transpose_convert_kernel(const float* __restrict__ W, _Float16* __restrict__ Wt,
                              int K, int N) {
  __shared__ float tile[64][65];
  const int k0 = blockIdx.x * 64;
  const int n0 = blockIdx.y * 64;
  const int tx = threadIdx.x & 15;   // col group (x4 floats)
  const int ty = threadIdx.x >> 4;   // row base
#pragma unroll
  for (int i = 0; i < 64; i += 16) {
    const float4 v = *(const float4*)&W[(size_t)(k0 + ty + i) * N + n0 + tx * 4];
    tile[ty + i][tx * 4 + 0] = v.x;
    tile[ty + i][tx * 4 + 1] = v.y;
    tile[ty + i][tx * 4 + 2] = v.z;
    tile[ty + i][tx * 4 + 3] = v.w;
  }
  __syncthreads();
#pragma unroll
  for (int i = 0; i < 64; i += 16) {
    const int nr = ty + i;     // output row (n)
    const int kc = tx * 4;     // output col (k)
    v4h pk;
#pragma unroll
    for (int j = 0; j < 4; ++j) pk[j] = (_Float16)tile[kc + j][nr];
    *(v4h*)&Wt[(size_t)(n0 + nr) * K + k0 + kc] = pk;
  }
}

// ---------------------------------------------------------------------------
// WMMA GEMM: C[M,N] = A[M,K](f16,row-major) * Bt[N,K]^T (f16, N-major)
// Block tile 128x128, 8 waves (4x2), wave tile 32x64 (2x4 wmma tiles), BK=32.
// Double-buffered LDS; next K-tile fetched with GLOBAL_LOAD_ASYNC_TO_LDS_B128
// (ASYNCcnt / s_wait_asynccnt) when available, else register-pipelined stores.
// MODE 0: store f16 row-major; MODE 1: store f16 transposed [N,M];
// MODE 2: store f32 row-major + bias.
// ---------------------------------------------------------------------------
template <int MODE>
__global__ __launch_bounds__(256)
void gemm_wmma_kernel(const _Float16* __restrict__ A, const _Float16* __restrict__ Bt,
                      _Float16* __restrict__ outH, float* __restrict__ outF,
                      const float* __restrict__ bias, int M, int N, int K) {
  constexpr int LS = 40;  // padded LDS row stride (f16 elems), 80B = 16B-aligned
  __shared__ _Float16 lA[2][128 * LS];
  __shared__ _Float16 lB[2][128 * LS];

  const int tid  = threadIdx.x;
  const int lane = tid & 31, wave = tid >> 5;
  const int lrow = lane & 15, hi = lane >> 4;
  const int wm = wave & 3;    // wave row (x32)
  const int wn = wave >> 2;   // wave col (x64)
  const int m0 = blockIdx.x * 128, n0 = blockIdx.y * 128;

  v8f acc[2][4] = {};

  const int lr = tid >> 2;          // 0..63
  const int lc = (tid & 3) << 3;    // 0,8,16,24

  const _Float16* gA0 = A  + (size_t)(m0 + lr)      * K + lc;
  const _Float16* gA1 = A  + (size_t)(m0 + lr + 64) * K + lc;
  const _Float16* gB0 = Bt + (size_t)(n0 + lr)      * K + lc;
  const _Float16* gB1 = Bt + (size_t)(n0 + lr + 64) * K + lc;

#if !USE_ASYNC_LDS
  v8h ra0, ra1, rb0, rb1;   // register pipeline for the sync fallback
  int pend = -1;            // pending buffer index
#endif

#if USE_ASYNC_LDS
#define TILE_PREFETCH(b, kk)                                                          \
  do {                                                                                \
    __builtin_amdgcn_global_load_async_to_lds_b128(                                   \
        (AS1 v4i*)(gA0 + (kk)), (AS3 v4i*)&lA[b][lr * LS + lc], 0, 0);                \
    __builtin_amdgcn_global_load_async_to_lds_b128(                                   \
        (AS1 v4i*)(gA1 + (kk)), (AS3 v4i*)&lA[b][(lr + 64) * LS + lc], 0, 0);         \
    __builtin_amdgcn_global_load_async_to_lds_b128(                                   \
        (AS1 v4i*)(gB0 + (kk)), (AS3 v4i*)&lB[b][lr * LS + lc], 0, 0);                \
    __builtin_amdgcn_global_load_async_to_lds_b128(                                   \
        (AS1 v4i*)(gB1 + (kk)), (AS3 v4i*)&lB[b][(lr + 64) * LS + lc], 0, 0);         \
  } while (0)
#define TILE_COMMIT()                                                                 \
  do {                                                                                \
    __builtin_amdgcn_s_wait_asynccnt(0);                                              \
    __syncthreads();                                                                  \
  } while (0)
#else
#define TILE_PREFETCH(b, kk)                                                          \
  do {                                                                                \
    ra0 = *(const v8h*)(gA0 + (kk));                                                  \
    ra1 = *(const v8h*)(gA1 + (kk));                                                  \
    rb0 = *(const v8h*)(gB0 + (kk));                                                  \
    rb1 = *(const v8h*)(gB1 + (kk));                                                  \
    pend = (b);                                                                       \
  } while (0)
#define TILE_COMMIT()                                                                 \
  do {                                                                                \
    *(v8h*)&lA[pend][lr * LS + lc]        = ra0;                                      \
    *(v8h*)&lA[pend][(lr + 64) * LS + lc] = ra1;                                      \
    *(v8h*)&lB[pend][lr * LS + lc]        = rb0;                                      \
    *(v8h*)&lB[pend][(lr + 64) * LS + lc] = rb1;                                      \
    __syncthreads();                                                                  \
  } while (0)
#endif

  auto compute = [&](int b) {
    v16h af[2], bf[4];
#pragma unroll
    for (int mt = 0; mt < 2; ++mt)
      af[mt] = afrag(&lA[b][(32 * wm + mt * 16) * LS], lrow, hi, LS);
#pragma unroll
    for (int nt = 0; nt < 4; ++nt)
      bf[nt] = bfrag(&lB[b][(64 * wn + nt * 16) * LS], lrow, hi, LS);
#pragma unroll
    for (int mt = 0; mt < 2; ++mt)
#pragma unroll
      for (int nt = 0; nt < 4; ++nt)
        acc[mt][nt] = __builtin_amdgcn_wmma_f32_16x16x32_f16(
            false, af[mt], false, bf[nt], (short)0, acc[mt][nt], false, false);
  };

  // prologue: stage tile 0
  TILE_PREFETCH(0, 0);
  TILE_COMMIT();

  for (int kk = 0; kk < K; kk += 64) {
    TILE_PREFETCH(1, kk + 32);   // kk+32 < K always (K multiple of 64)
    compute(0);
    TILE_COMMIT();
    if (kk + 64 < K) {
      TILE_PREFETCH(0, kk + 64);
      compute(1);
      TILE_COMMIT();
    } else {
      compute(1);
    }
  }
#undef TILE_PREFETCH
#undef TILE_COMMIT

  // store (C layout: lane = col lrow, VGPR r -> row r + 8*hi)
#pragma unroll
  for (int mt = 0; mt < 2; ++mt) {
#pragma unroll
    for (int nt = 0; nt < 4; ++nt) {
      const int col  = n0 + 64 * wn + nt * 16 + lrow;
      const int rowb = m0 + 32 * wm + mt * 16 + 8 * hi;
      if constexpr (MODE == 0) {
#pragma unroll
        for (int r = 0; r < 8; ++r)
          outH[(size_t)(rowb + r) * N + col] = (_Float16)acc[mt][nt][r];
      } else if constexpr (MODE == 1) {
        v8h pk;
#pragma unroll
        for (int r = 0; r < 8; ++r) pk[r] = (_Float16)acc[mt][nt][r];
        *(v8h*)&outH[(size_t)col * M + rowb] = pk;   // contiguous 16B transposed store
      } else {
        const float bb = bias[col];
#pragma unroll
        for (int r = 0; r < 8; ++r)
          outF[(size_t)(rowb + r) * N + col] = acc[mt][nt][r] + bb;
      }
    }
  }
}

// ---------------------------------------------------------------------------
// Flash attention (online softmax). One block = 128 query rows of one (b,h);
// wave w owns 16 query rows. Q fragments register-resident. Key blocks of 32.
// Qh/Kh: [4096, 2048] f16 (token-major, col = h*128+d).
// Vt:    [2048, 4096] f16 (d-major, col = token)  -> contiguous B fragments.
// Oh:    [4096, 2048] f16 output (A-matrix for final projection).
// ---------------------------------------------------------------------------
__global__ __launch_bounds__(256)
void flash_attn_kernel(const _Float16* __restrict__ Qh, const _Float16* __restrict__ Kh,
                       const _Float16* __restrict__ Vt, _Float16* __restrict__ Oh) {
  __shared__ _Float16 pbuf[8 * 16 * 32];   // per-wave P-tile repack scratch
  const int tid  = threadIdx.x;
  const int lane = tid & 31, wave = tid >> 5;
  const int lrow = lane & 15, hi = lane >> 4;
  const int b = blockIdx.y >> 4;
  const int h = blockIdx.y & 15;
  const int q0 = blockIdx.x * 128 + wave * 16;

  const _Float16* Qb = Qh + (size_t)(b * SEQ) * EMBED + h * HDIM;
  const _Float16* Kb = Kh + (size_t)(b * SEQ) * EMBED + h * HDIM;
  const _Float16* Vb = Vt + (size_t)(h * HDIM) * MTOT + b * SEQ;
  _Float16* pw = &pbuf[wave * 16 * 32];

  // Q fragments: 4 K-slabs of 32 over d=128
  v16h qf[4];
#pragma unroll
  for (int s = 0; s < 4; ++s) {
    const _Float16* p = Qb + (size_t)(q0 + lrow) * EMBED + s * 32;
    qf[s] = pack16(*(const v8h*)(p + 8 * hi), *(const v8h*)(p + 16 + 8 * hi));
  }

  v8f o[8] = {};
  float mrun[8], lrun[8];
#pragma unroll
  for (int r = 0; r < 8; ++r) { mrun[r] = -1e30f; lrun[r] = 0.0f; }

  const float C = 0.08838834764831845f * 1.4426950408889634f;  // scale * log2(e)

  for (int kb = 0; kb < SEQ; kb += 32) {
    // scores: [16q x 32k] in log2 domain, 2 n-tiles
    v8f s0 = {}, s1 = {};
#pragma unroll
    for (int s = 0; s < 4; ++s) {
      const _Float16* p0 = Kb + (size_t)(kb + lrow) * EMBED + s * 32;
      const _Float16* p1 = Kb + (size_t)(kb + 16 + lrow) * EMBED + s * 32;
      v16h k0 = pack16(*(const v8h*)(p0 + 8 * hi), *(const v8h*)(p0 + 16 + 8 * hi));
      v16h k1 = pack16(*(const v8h*)(p1 + 8 * hi), *(const v8h*)(p1 + 16 + 8 * hi));
      s0 = __builtin_amdgcn_wmma_f32_16x16x32_f16(false, qf[s], false, k0, (short)0, s0, false, false);
      s1 = __builtin_amdgcn_wmma_f32_16x16x32_f16(false, qf[s], false, k1, (short)0, s1, false, false);
    }

    // online softmax: per-row (r + 8*hi) stats; rows live across the 16-lane group
    float alpha[8];
#pragma unroll
    for (int r = 0; r < 8; ++r) {
      float a = s0[r] * C, c = s1[r] * C;
      float mx = fmaxf(a, c);
#pragma unroll
      for (int off = 1; off < 16; off <<= 1)
        mx = fmaxf(mx, __shfl_xor(mx, off, 32));
      const float mnew = fmaxf(mrun[r], mx);
      alpha[r] = exp2f(mrun[r] - mnew);
      mrun[r] = mnew;
      const float p0 = exp2f(a - mnew);
      const float p1 = exp2f(c - mnew);
      s0[r] = p0; s1[r] = p1;
      float rs = p0 + p1;
#pragma unroll
      for (int off = 1; off < 16; off <<= 1)
        rs += __shfl_xor(rs, off, 32);
      lrun[r] = lrun[r] * alpha[r] + rs;
    }
#pragma unroll
    for (int j = 0; j < 8; ++j)
#pragma unroll
      for (int r = 0; r < 8; ++r) o[j][r] *= alpha[r];

    // repack P: C layout -> A-fragment layout via per-wave LDS (DS pipe is in-order per wave)
#pragma unroll
    for (int r = 0; r < 8; ++r) {
      pw[(r + 8 * hi) * 32 + lrow]      = (_Float16)s0[r];
      pw[(r + 8 * hi) * 32 + 16 + lrow] = (_Float16)s1[r];
    }
    const v16h pf = pack16(*(const v8h*)&pw[lrow * 32 + 8 * hi],
                           *(const v8h*)&pw[lrow * 32 + 16 + 8 * hi]);

    // O += P x V : 8 d-tiles of 16
#pragma unroll
    for (int j = 0; j < 8; ++j) {
      const _Float16* vp = Vb + (size_t)(j * 16 + lrow) * MTOT + kb;
      v16h vf = pack16(*(const v8h*)(vp + 8 * hi), *(const v8h*)(vp + 16 + 8 * hi));
      o[j] = __builtin_amdgcn_wmma_f32_16x16x32_f16(false, pf, false, vf, (short)0, o[j], false, false);
    }
  }

  // finalize: O /= l, store f16 [token, h*128+d]
  float rl[8];
#pragma unroll
  for (int r = 0; r < 8; ++r) rl[r] = 1.0f / lrun[r];
#pragma unroll
  for (int j = 0; j < 8; ++j) {
    const int col = h * HDIM + j * 16 + lrow;
#pragma unroll
    for (int r = 0; r < 8; ++r) {
      const int row = b * SEQ + q0 + r + 8 * hi;
      Oh[(size_t)row * EMBED + col] = (_Float16)(o[j][r] * rl[r]);
    }
  }
}

// ---------------------------------------------------------------------------
extern "C" void kernel_launch(void* const* d_in, const int* in_sizes, int n_in,
                              void* d_out, int out_size, void* d_ws, size_t ws_size,
                              hipStream_t stream) {
  (void)in_sizes; (void)n_in; (void)out_size; (void)ws_size;
  const float* x  = (const float*)d_in[0];
  const float* Wq = (const float*)d_in[1];
  const float* Wk = (const float*)d_in[2];
  const float* Wv = (const float*)d_in[3];
  const float* Wo = (const float*)d_in[4];
  const float* bo = (const float*)d_in[5];
  float* out = (float*)d_out;

  const size_t SZ_X = (size_t)MTOT * EMBED;    // 8M elems
  const size_t SZ_W = (size_t)EMBED * EMBED;   // 4M elems

  _Float16* xh  = (_Float16*)d_ws;
  _Float16* Wqt = xh  + SZ_X;
  _Float16* Wkt = Wqt + SZ_W;
  _Float16* Wvt = Wkt + SZ_W;
  _Float16* Wot = Wvt + SZ_W;
  _Float16* Qh  = Wot + SZ_W;
  _Float16* Kh  = Qh  + SZ_X;
  _Float16* Vth = Kh  + SZ_X;
  _Float16* Oh  = Vth + SZ_X;   // total ~112MB f16 workspace

  convert_f32_f16_kernel<<<(int)(SZ_X / (256 * 4)), 256, 0, stream>>>(x, xh, (int)SZ_X);

  dim3 tg(EMBED / 64, EMBED / 64);
  transpose_convert_kernel<<<tg, 256, 0, stream>>>(Wq, Wqt, EMBED, EMBED);
  transpose_convert_kernel<<<tg, 256, 0, stream>>>(Wk, Wkt, EMBED, EMBED);
  transpose_convert_kernel<<<tg, 256, 0, stream>>>(Wv, Wvt, EMBED, EMBED);
  transpose_convert_kernel<<<tg, 256, 0, stream>>>(Wo, Wot, EMBED, EMBED);

  dim3 gg(MTOT / 128, EMBED / 128);  // 32 x 16 blocks
  gemm_wmma_kernel<0><<<gg, 256, 0, stream>>>(xh, Wqt, Qh, nullptr, nullptr, MTOT, EMBED, EMBED);
  gemm_wmma_kernel<0><<<gg, 256, 0, stream>>>(xh, Wkt, Kh, nullptr, nullptr, MTOT, EMBED, EMBED);
  gemm_wmma_kernel<1><<<gg, 256, 0, stream>>>(xh, Wvt, Vth, nullptr, nullptr, MTOT, EMBED, EMBED);

  flash_attn_kernel<<<dim3(SEQ / 128, BATCH * HEADS), 256, 0, stream>>>(Qh, Kh, Vth, Oh);

  gemm_wmma_kernel<2><<<gg, 256, 0, stream>>>(Oh, Wot, nullptr, out, bo, MTOT, EMBED, EMBED);
}
